// VarianceAdaptor_23605140259247
// MI455X (gfx1250) — matrix operand order repaired and checked by reading, not compiled, not run on hardware
//
#include <hip/hip_runtime.h>
#include <hip/hip_bf16.h>

typedef __attribute__((ext_vector_type(16))) __bf16 v16bf;
typedef __attribute__((ext_vector_type(8)))  __bf16 v8bf;
typedef __attribute__((ext_vector_type(8)))  float  v8f;
typedef __attribute__((ext_vector_type(4)))  int    v4i;

constexpr int kB = 32, kS = 512, kD = 512, kF = 512, kT = 4096, kNB = 256;
constexpr int kRows = kB * kS;           // 16384
constexpr int kSP   = kS + 2;            // padded sequence length (1 zero row each side)
constexpr int TM = 128, TN = 128, TK = 32;
constexpr int kSteps = 3 * (kD / TK);    // 3 taps x 16 k-steps = 48

#if defined(__HIP_DEVICE_COMPILE__) && __has_builtin(__builtin_amdgcn_global_load_async_to_lds_b128)
#define HAVE_ASYNC 1
#else
#define HAVE_ASYNC 0
#endif

union V16 { v16bf v; v8bf h[2]; };

__device__ __forceinline__ void cp16_to_lds(void* ldsDst, const void* gSrc) {
#if HAVE_ASYNC
  void* g = const_cast<void*>(gSrc);
  __builtin_amdgcn_global_load_async_to_lds_b128(
      (__attribute__((address_space(1))) v4i*)g,
      (__attribute__((address_space(3))) v4i*)ldsDst, 0, 0);
#else
  *(float4*)ldsDst = *(const float4*)gSrc;
#endif
}

// retire everything except the newest 4 async copies (next-buffer prefetch)
__device__ __forceinline__ void async_wait_keep4() {
#if HAVE_ASYNC
#if __has_builtin(__builtin_amdgcn_s_wait_asynccnt)
  __builtin_amdgcn_s_wait_asynccnt(4);
#else
  asm volatile("s_wait_asynccnt 4" ::: "memory");
#endif
#endif
}
__device__ __forceinline__ void async_wait_all() {
#if HAVE_ASYNC
#if __has_builtin(__builtin_amdgcn_s_wait_asynccnt)
  __builtin_amdgcn_s_wait_asynccnt(0);
#else
  asm volatile("s_wait_asynccnt 0" ::: "memory");
#endif
#endif
}

// ---------------------------------------------------------------------------
// K=3 "same" conv1d as 3 shifted GEMMs over a zero-padded activation buffer:
//   Y[r,f] = sum_tap sum_d Ap[pad(r)+tap-1, d] * WT[tap, f, d]
// Ap: [kB][kSP][kD] bf16, rows 0 and kSP-1 of each batch are zeros.
// Padded row for (global row gr, tap): gr + 2*(gr/kS) + tap  (block-uniform shift)
// WT: [3][kF][kD] bf16 (pre-transposed). Y: [kRows][kF] fp32.
// Double-buffered LDS; async copies for step n+1 overlap WMMAs of step n.
// ---------------------------------------------------------------------------
__global__ __launch_bounds__(256)
void conv_gemm(const __bf16* __restrict__ Ap, const __bf16* __restrict__ WT,
               float* __restrict__ Y)
{
  __shared__ __align__(16) __bf16 lA[2][TM * TK];
  __shared__ __align__(16) __bf16 lB[2][TN * TK];

  const int tid  = threadIdx.x;
  const int lane = tid & 31;
  const int wave = tid >> 5;
  const int l    = lane & 15;
  const int hi   = lane >> 4;       // half-wave
  const int wm   = wave >> 1;       // 0..3 (M)
  const int wn   = wave & 1;        // 0..1 (N)
  const int row0 = blockIdx.x * TM;
  const int col0 = blockIdx.y * TN;
  const int bidx = row0 >> 9;       // batch of this row tile (TM divides kS)

  // per-thread staging coordinates (step-invariant)
  const int e0 = tid * 8;           // pass 0 element
  const int e1 = (256 + tid) * 8;   // pass 1 element
  const int r0 = e0 >> 5, kk0 = e0 & 31;
  const int r1 = e1 >> 5, kk1 = e1 & 31;

  v8f acc[2][4];
#pragma unroll
  for (int mi = 0; mi < 2; ++mi)
#pragma unroll
    for (int ni = 0; ni < 4; ++ni)
#pragma unroll
      for (int j = 0; j < 8; ++j) acc[mi][ni][j] = 0.0f;

  // issue the 4 async copies (2 A-chunks + 2 B-chunks) for `step` into buffer `buf`
  auto stage = [&](int step, int buf) {
    const int tap = step >> 4;              // 0..2
    const int k0  = (step & 15) * TK;       // 0..480
    const __bf16* As = Ap + (size_t)(row0 + 2 * bidx + tap) * kD + k0;
    const __bf16* Ws = WT + ((size_t)tap * kF + col0) * kD + k0;
    cp16_to_lds(&lA[buf][r0 * TK + kk0], As + (size_t)r0 * kD + kk0);
    cp16_to_lds(&lA[buf][r1 * TK + kk1], As + (size_t)r1 * kD + kk1);
    cp16_to_lds(&lB[buf][r0 * TK + kk0], Ws + (size_t)r0 * kD + kk0);
    cp16_to_lds(&lB[buf][r1 * TK + kk1], Ws + (size_t)r1 * kD + kk1);
  };

  stage(0, 0);
  for (int step = 0; step < kSteps; ++step) {
    const int cur = step & 1;
    __syncthreads();                        // everyone done reading buffer cur^1
    if (step + 1 < kSteps) {
      stage(step + 1, cur ^ 1);             // prefetch next step
      async_wait_keep4();                   // retire current buffer's copies only
    } else {
      async_wait_all();
    }
    __syncthreads();                        // current buffer visible to all waves

    // ---- A fragments (ISA 16x32 bf16: lanes<16 K0-7/16-23, lanes>=16 K8-15/24-31)
    v16bf af[2];
#pragma unroll
    for (int mi = 0; mi < 2; ++mi) {
      int r = wm * 32 + mi * 16 + l;
      const v8bf* pa = (const v8bf*)&lA[cur][r * TK + hi * 8];
      V16 t; t.h[0] = pa[0]; t.h[1] = pa[2];   // +16 bf16 = +2 v8bf
      af[mi] = t.v;
    }
    // ---- B fragments (lane n = lane&15; lanes<16 K0-7/16-23, lanes>=16 K8-15/24-31)
    v16bf bfr[4];
#pragma unroll
    for (int ni = 0; ni < 4; ++ni) {
      int n = wn * 64 + ni * 16 + l;
      const v8bf* pb = (const v8bf*)&lB[cur][n * TK + hi * 8];
      V16 t; t.h[0] = pb[0]; t.h[1] = pb[2];
      bfr[ni] = t.v;
    }
#pragma unroll
    for (int mi = 0; mi < 2; ++mi)
#pragma unroll
      for (int ni = 0; ni < 4; ++ni)
        acc[mi][ni] = __builtin_amdgcn_wmma_f32_16x16x32_bf16(
            false, af[mi], false, bfr[ni], (short)0, acc[mi][ni], false, false);
  }

  // C/D layout: VGPR j -> (M = hi*8 + j, N = l)
#pragma unroll
  for (int mi = 0; mi < 2; ++mi) {
#pragma unroll
    for (int ni = 0; ni < 4; ++ni) {
      int col = col0 + wn * 64 + ni * 16 + l;
#pragma unroll
      for (int j = 0; j < 8; ++j) {
        int row = row0 + wm * 32 + mi * 16 + hi * 8 + j;
        Y[(size_t)row * kF + col] = acc[mi][ni][j];
      }
    }
  }
}

__device__ __forceinline__ size_t pad_row(int row) {
  return (size_t)(row + 2 * (row >> 9) + 1) * kD;   // index into [kB][kSP][kD]
}

// ---------------------------------------------------------------------------
// h = layernorm(relu(Y + bias)) * g + be -> fp32 (Hf) and padded bf16 (Hbp)
// ---------------------------------------------------------------------------
__global__ __launch_bounds__(128)
void relu_ln(const float* __restrict__ Y, const float* __restrict__ bias,
             const float* __restrict__ g, const float* __restrict__ be,
             float* __restrict__ Hf, __bf16* __restrict__ Hbp)
{
  const int row = blockIdx.x;
  const int tid = threadIdx.x;
  const int lane = tid & 31, wave = tid >> 5;
  float4 v = ((const float4*)(Y + (size_t)row * kF))[tid];
  float4 bb = ((const float4*)bias)[tid];
  v.x = fmaxf(v.x + bb.x, 0.f);
  v.y = fmaxf(v.y + bb.y, 0.f);
  v.z = fmaxf(v.z + bb.z, 0.f);
  v.w = fmaxf(v.w + bb.w, 0.f);
  float s  = v.x + v.y + v.z + v.w;
  float ss = v.x * v.x + v.y * v.y + v.z * v.z + v.w * v.w;
#pragma unroll
  for (int m = 16; m >= 1; m >>= 1) { s += __shfl_xor(s, m); ss += __shfl_xor(ss, m); }
  __shared__ float rs[4], rss[4];
  if (lane == 0) { rs[wave] = s; rss[wave] = ss; }
  __syncthreads();
  s  = rs[0] + rs[1] + rs[2] + rs[3];
  ss = rss[0] + rss[1] + rss[2] + rss[3];
  const float mean = s * (1.0f / kF);
  const float var  = ss * (1.0f / kF) - mean * mean;
  const float rstd = rsqrtf(var + 1e-5f);
  float4 gg = ((const float4*)g)[tid];
  float4 bz = ((const float4*)be)[tid];
  float4 o;
  o.x = (v.x - mean) * rstd * gg.x + bz.x;
  o.y = (v.y - mean) * rstd * gg.y + bz.y;
  o.z = (v.z - mean) * rstd * gg.z + bz.z;
  o.w = (v.w - mean) * rstd * gg.w + bz.w;
  ((float4*)(Hf + (size_t)row * kF))[tid] = o;
  __bf16* hb = Hbp + pad_row(row) + tid * 4;
  hb[0] = (__bf16)o.x; hb[1] = (__bf16)o.y; hb[2] = (__bf16)o.z; hb[3] = (__bf16)o.w;
}

// out[r] = mask[r] ? 0 : dot(H[r,:], wl) + bl
__global__ __launch_bounds__(128)
void linear_mask(const float* __restrict__ H, const float* __restrict__ wl,
                 const float* __restrict__ bl, const unsigned char* __restrict__ mask,
                 float* __restrict__ outp)
{
  const int row = blockIdx.x;
  const int tid = threadIdx.x;
  const int lane = tid & 31, wave = tid >> 5;
  float4 h = ((const float4*)(H + (size_t)row * kF))[tid];
  float4 w = ((const float4*)wl)[tid];
  float s = h.x * w.x + h.y * w.y + h.z * w.z + h.w * w.w;
#pragma unroll
  for (int m = 16; m >= 1; m >>= 1) s += __shfl_xor(s, m);
  __shared__ float rs[4];
  if (lane == 0) rs[wave] = s;
  __syncthreads();
  if (tid == 0) {
    float t = rs[0] + rs[1] + rs[2] + rs[3] + bl[0];
    outp[row] = mask[row] ? 0.0f : t;
  }
}

// x_cur[row,:] += emb[searchsorted_left(bins, truth[row]), :]; refresh padded bf16
__global__ __launch_bounds__(128)
void emb_add(float* __restrict__ xc, __bf16* __restrict__ xbp,
             const float* __restrict__ truth, const float* __restrict__ bins,
             const float* __restrict__ emb)
{
  const int row = blockIdx.x, tid = threadIdx.x;
  const float t = truth[row];
  int lo = 0, hiB = kNB - 1;            // 255 bin edges
  while (lo < hiB) { int mid = (lo + hiB) >> 1; if (bins[mid] < t) lo = mid + 1; else hiB = mid; }
  float4 e = ((const float4*)(emb + (size_t)lo * kD))[tid];
  float4* xp = (float4*)(xc + (size_t)row * kD);
  float4 v = xp[tid];
  v.x += e.x; v.y += e.y; v.z += e.z; v.w += e.w;
  xp[tid] = v;
  __bf16* xb = xbp + pad_row(row) + tid * 4;
  xb[0] = (__bf16)v.x; xb[1] = (__bf16)v.y; xb[2] = (__bf16)v.z; xb[3] = (__bf16)v.w;
}

__global__ __launch_bounds__(128)
void init_cast(const float* __restrict__ x, float* __restrict__ xc, __bf16* __restrict__ xbp)
{
  const int row = blockIdx.x, tid = threadIdx.x;
  float4 v = ((const float4*)(x + (size_t)row * kD))[tid];
  ((float4*)(xc + (size_t)row * kD))[tid] = v;
  __bf16* p = xbp + pad_row(row) + tid * 4;
  p[0] = (__bf16)v.x; p[1] = (__bf16)v.y; p[2] = (__bf16)v.z; p[3] = (__bf16)v.w;
}

// zero the 2 pad rows of each batch in both padded bf16 buffers
__global__ __launch_bounds__(128)
void zero_pads(__bf16* __restrict__ xbp, __bf16* __restrict__ hbp)
{
  const int b = blockIdx.x >> 1;
  const int which = blockIdx.x & 1;                  // 0 = front pad, 1 = back pad
  const size_t pr = ((size_t)b * kSP + (which ? kSP - 1 : 0)) * kD;
  const int tid = threadIdx.x;
  __bf16 z = (__bf16)0.0f;
  __bf16* p0 = xbp + pr + tid * 4;
  __bf16* p1 = hbp + pr + tid * 4;
#pragma unroll
  for (int j = 0; j < 4; ++j) { p0[j] = z; p1[j] = z; }
}

// w[pred][tap][d][f] (fp32) -> wT[pred][tap][f][d] (bf16)
__global__ __launch_bounds__(256)
void wcast(const float* __restrict__ w, __bf16* __restrict__ wT, int Cin)
{
  size_t idx = (size_t)blockIdx.x * 256 + threadIdx.x;
  size_t total = (size_t)9 * Cin * kF;
  if (idx >= total) return;
  int d = (int)(idx % Cin);
  size_t r = idx / Cin;
  int f = (int)(r % kF);
  size_t pk = r / kF;                   // pred*3 + tap
  wT[idx] = (__bf16)w[(pk * Cin + d) * kF + f];
}

__global__ __launch_bounds__(512)
void cumsum_k(const int* __restrict__ dur, int* __restrict__ cum, float* __restrict__ mel_len)
{
  const int b = blockIdx.x, tid = threadIdx.x;
  __shared__ int buf[kS];
  buf[tid] = dur[b * kS + tid];
  __syncthreads();
  for (int off = 1; off < kS; off <<= 1) {
    int v = (tid >= off) ? buf[tid - off] : 0;
    __syncthreads();
    buf[tid] += v;
    __syncthreads();
  }
  cum[b * kS + tid] = buf[tid];
  if (tid == kS - 1) mel_len[b] = (float)buf[tid];
}

__global__ __launch_bounds__(128)
void length_regulate(const float* __restrict__ xc, const int* __restrict__ cum,
                     float* __restrict__ xout)
{
  const int bt = blockIdx.x;
  const int b = bt >> 12;               // / kT
  const int t = bt & (kT - 1);
  const int tid = threadIdx.x;
  const int* c = cum + b * kS;
  int lo = 0, hiS = kS;                 // searchsorted 'right'
  while (lo < hiS) { int mid = (lo + hiS) >> 1; if (c[mid] <= t) lo = mid + 1; else hiS = mid; }
  int idx = lo < kS - 1 ? lo : kS - 1;
  bool valid = t < c[kS - 1];
  float4 v = valid ? ((const float4*)(xc + ((size_t)b * kS + idx) * kD))[tid]
                   : make_float4(0.f, 0.f, 0.f, 0.f);
  ((float4*)(xout + (size_t)bt * kD))[tid] = v;
}

__global__ __launch_bounds__(256)
void mask_copy(const unsigned char* __restrict__ m, float* __restrict__ o, int n)
{
  int i = blockIdx.x * 256 + threadIdx.x;
  if (i < n) o[i] = m[i] ? 1.0f : 0.0f;
}

extern "C" void kernel_launch(void* const* d_in, const int* in_sizes, int n_in,
                              void* d_out, int out_size, void* d_ws, size_t ws_size,
                              hipStream_t stream)
{
  const float* x              = (const float*)d_in[0];
  const unsigned char* src_mask = (const unsigned char*)d_in[1];
  const unsigned char* mel_mask = (const unsigned char*)d_in[2];
  const float* pitch_truth    = (const float*)d_in[3];
  const float* energy_truth   = (const float*)d_in[4];
  const int*   dur            = (const int*)d_in[5];
  const float* conv1_w = (const float*)d_in[6];
  const float* conv1_b = (const float*)d_in[7];
  const float* ln1_g   = (const float*)d_in[8];
  const float* ln1_b   = (const float*)d_in[9];
  const float* conv2_w = (const float*)d_in[10];
  const float* conv2_b = (const float*)d_in[11];
  const float* ln2_g   = (const float*)d_in[12];
  const float* ln2_b   = (const float*)d_in[13];
  const float* lin_w   = (const float*)d_in[14];
  const float* lin_b   = (const float*)d_in[15];
  const float* pitch_emb   = (const float*)d_in[16];
  const float* energy_emb  = (const float*)d_in[17];
  const float* pitch_bins  = (const float*)d_in[18];
  const float* energy_bins = (const float*)d_in[19];

  char* ws = (char*)d_ws;
  float*  x_cur = (float*)ws;  ws += (size_t)kRows * kD * 4;
  __bf16* x_bp  = (__bf16*)ws; ws += (size_t)kB * kSP * kD * 2;
  __bf16* h_bp  = (__bf16*)ws; ws += (size_t)kB * kSP * kD * 2;
  float*  Yf    = (float*)ws;  ws += (size_t)kRows * kF * 4;
  float*  Hf    = (float*)ws;  ws += (size_t)kRows * kF * 4;
  __bf16* w1T   = (__bf16*)ws; ws += (size_t)9 * kD * kF * 2;
  __bf16* w2T   = (__bf16*)ws; ws += (size_t)9 * kF * kF * 2;
  int*    cum   = (int*)ws;    ws += (size_t)kRows * 4;

  float* outF     = (float*)d_out;
  float* o_xout   = outF;
  float* o_pitch  = outF + (size_t)kB * kT * kD;
  float* o_energy = o_pitch + kRows;
  float* o_dur    = o_energy + kRows;
  float* o_mellen = o_dur + kRows;
  float* o_mmask  = o_mellen + kB;

  init_cast<<<kRows, 128, 0, stream>>>(x, x_cur, x_bp);
  zero_pads<<<2 * kB, 128, 0, stream>>>(x_bp, h_bp);
  wcast<<<(9 * kD * kF + 255) / 256, 256, 0, stream>>>(conv1_w, w1T, kD);
  wcast<<<(9 * kF * kF + 255) / 256, 256, 0, stream>>>(conv2_w, w2T, kF);

  dim3 gg(kRows / TM, kF / TN);
  const float* truths[2] = {pitch_truth, energy_truth};
  const float* binses[2] = {pitch_bins, energy_bins};
  const float* embs[2]   = {pitch_emb, energy_emb};
  float* preds[3] = {o_pitch, o_energy, o_dur};

  for (int i = 0; i < 3; ++i) {
    conv_gemm<<<gg, 256, 0, stream>>>(x_bp, w1T + (size_t)i * 3 * kD * kF, Yf);
    relu_ln<<<kRows, 128, 0, stream>>>(Yf, conv1_b + i * kF, ln1_g + i * kF,
                                       ln1_b + i * kF, Hf, h_bp);
    conv_gemm<<<gg, 256, 0, stream>>>(h_bp, w2T + (size_t)i * 3 * kF * kF, Yf);
    relu_ln<<<kRows, 128, 0, stream>>>(Yf, conv2_b + i * kF, ln2_g + i * kF,
                                       ln2_b + i * kF, Hf, h_bp);
    linear_mask<<<kRows, 128, 0, stream>>>(Hf, lin_w + (size_t)i * kF, lin_b + i,
                                           src_mask, preds[i]);
    if (i < 2)
      emb_add<<<kRows, 128, 0, stream>>>(x_cur, x_bp, truths[i], binses[i], embs[i]);
  }

  cumsum_k<<<kB, kS, 0, stream>>>(dur, cum, o_mellen);
  length_regulate<<<kB * kT, 128, 0, stream>>>(x_cur, cum, o_xout);
  mask_copy<<<(kB * kT + 255) / 256, 256, 0, stream>>>(mel_mask, o_mmask, kB * kT);
}
